// FuncSelfAttention_70085276336129
// MI455X (gfx1250) — matrix operand order repaired and checked by reading, not compiled, not run on hardware
//
#include <hip/hip_runtime.h>
#include <math.h>

// ---------- problem constants (fixed by setup_inputs) ----------
#define BB   4
#define SS   64
#define BS   256      // B*S
#define CC   64
#define NQKV 192      // 3*C
#define PIX  1024     // H*W = 32*32
#define MD   16       // MODES
#define NMODE 256     // MD*MD
#define NH   8
#define HD   8
#define KATT 8192     // HD*PIX
#define ANG32 0.19634954084936207f   // 2*pi/32
#define NFRAG 15

typedef __attribute__((ext_vector_type(16))) __bf16 v16bf;
typedef __attribute__((ext_vector_type(8)))  float  v8f;
typedef __attribute__((ext_vector_type(8)))  unsigned short us8;

// ---------- bf16 bit helpers (round-to-nearest-even) ----------
__device__ __forceinline__ unsigned short f2bf(float f) {
  unsigned u = __float_as_uint(f);
  return (unsigned short)((u + 0x7FFFu + ((u >> 16) & 1u)) >> 16);
}
__device__ __forceinline__ __bf16 bfbits(unsigned short s) {
  __bf16 b; __builtin_memcpy(&b, &s, 2); return b;
}
__device__ __forceinline__ v16bf mkbf(us8 lo, us8 hi) {
  v16bf v;
  __builtin_memcpy(&v, &lo, 16);
  __builtin_memcpy(((char*)&v) + 16, &hi, 16);
  return v;
}
__device__ __forceinline__ us8 pack8(const v8f& c) {
  us8 r;
#pragma unroll
  for (int i = 0; i < 8; ++i) r[i] = f2bf(c[i]);
  return r;
}

// WMMA 16x16x32 bf16 lane layouts (CDNA5 ISA §7.12.2):
// A (16x32, MxK): lane L -> row M=L&15, half=L>>4; elems 0..15: K=half*16+0..15 (this file's order, not AMD's)
// B (32x16, KxN): lane L -> col N=L&15; elems 0..15: K=half*16+0..15 (this file's order, not AMD's)
// D (16x16 f32):  lane L -> col N=L&15; elem r -> row M=r+8*half
__device__ __forceinline__ int a_kpos(int j, int half) {
  return 2 * j + half * 16;
}

// Pre-built twiddle fragments, stored as [frag][lane][16] bf16 (32B per lane).
// frag 0/1:  dft_x  B: cos / -sin        (col kx=l15, K=x)
// frag 2/3/4: dft_y A: cos / sin / -sin  (row ky=l15, K=y)
// frag 5..10: idft E (A-layout): {cos,sin,-sin} x {y0=0, y0=16} (K=ky, padded)
// frag 11..14: idft C (B-layout): {w*cos, -w*sin} x {x0=0, x0=16} (K=kx, padded)
__device__ __forceinline__ v16bf load_frag(const unsigned short* __restrict__ tw,
                                           int f, int lane) {
  const unsigned short* p = tw + (((size_t)f * 32 + lane) << 4);
  return mkbf(*(const us8*)p, *(const us8*)(p + 8));
}

// =====================================================================
// Kernel T: build the twiddle-fragment table (runs once per launch; tiny)
// =====================================================================
__global__ void twiddle_init_kernel(unsigned short* __restrict__ tw) {
  int id = blockIdx.x * 256 + threadIdx.x;     // < 15*512
  if (id >= NFRAG * 512) return;
  int f = id >> 9;
  int rem = id & 511;
  int lane = rem >> 4;
  int q = rem & 15;
  int half = lane >> 4;
  int l15 = lane & 15;
  float val = 0.f;
  if (f <= 1) {                       // dft_x B-layout
    int x = half * 16 + q;
    float ang = (float)((l15 * x) & 31) * ANG32;
    val = (f == 0) ? cosf(ang) : -sinf(ang);
  } else if (f <= 4) {                // dft_y A-layout
    int y = a_kpos(q >> 1, half) + (q & 1);
    float ang = (float)((l15 * y) & 31) * ANG32;
    float s = sinf(ang);
    val = (f == 2) ? cosf(ang) : ((f == 3) ? s : -s);
  } else if (f <= 10) {               // idft E, A-layout, K padded 16->32
    int y0 = (f >= 8) ? 16 : 0;
    int fi = (f - 5) % 3;
    int K = a_kpos(q >> 1, half) + (q & 1);
    if (K < MD) {
      float ang = (float)((K * (y0 + l15)) & 31) * ANG32;
      float s = sinf(ang);
      val = (fi == 0) ? cosf(ang) : ((fi == 1) ? s : -s);
    }
  } else {                            // idft C, B-layout, K padded, weights folded
    int x0 = (f >= 13) ? 16 : 0;
    int fi = (f - 11) & 1;
    int K = half * 16 + q;
    if (K < MD) {
      float wgt = (K == 0) ? (1.0f / 1024.0f) : (2.0f / 1024.0f);
      float ang = (float)((K * (x0 + l15)) & 31) * ANG32;
      val = (fi == 0) ? wgt * cosf(ang) : -wgt * sinf(ang);
    }
  }
  tw[id] = f2bf(val);
}

// =====================================================================
// Kernel 0: f32 -> bf16 elementwise (seq)
// =====================================================================
__global__ void cvt_bf16_kernel(const float* __restrict__ src, unsigned short* __restrict__ dst) {
  size_t id = (size_t)blockIdx.x * 256 + threadIdx.x;
  dst[id] = f2bf(src[id]);
}

// =====================================================================
// Kernel F1: x-direction DFT as one WMMA GEMM.
//   src: bf16 rows [(BS*64*32) x 32]; out: Ut[plane][kx*32 + y] bf16
// =====================================================================
__global__ void dft_x_kernel(const unsigned short* __restrict__ src,
                             const unsigned short* __restrict__ tw,
                             unsigned short* __restrict__ Utr,
                             unsigned short* __restrict__ Uti) {
  const int lane = threadIdx.x & 31;
  const int wid  = (int)((blockIdx.x * 256 + threadIdx.x) >> 5);  // m-tile, < 32768
  const int half = lane >> 4;
  const int kx   = lane & 15;

  const unsigned short* ap = src + ((size_t)wid * 16 + (lane & 15)) * 32;
  v16bf A = mkbf(*(const us8*)(ap + half * 16), *(const us8*)(ap + half * 16 + 8));
  v16bf Bc  = load_frag(tw, 0, lane);
  v16bf Bsn = load_frag(tw, 1, lane);

  v8f zero = {};
  v8f ur = __builtin_amdgcn_wmma_f32_16x16x32_bf16(false, A, false, Bc,  (short)0, zero, false, false);
  v8f ui = __builtin_amdgcn_wmma_f32_16x16x32_bf16(false, A, false, Bsn, (short)0, zero, false, false);

  const int plane = wid >> 1;
  const int yb = ((wid & 1) << 4) + half * 8;
  us8 pr = pack8(ur), pi = pack8(ui);
  *(us8*)(Utr + (size_t)plane * 512 + kx * 32 + yb) = pr;
  *(us8*)(Uti + (size_t)plane * 512 + kx * 32 + yb) = pi;
}

// =====================================================================
// Kernel F2: y-direction DFT, one wave per plane (16384 planes).
//   out: mode-major A operands Are/Aim [256][BS][64] bf16
// =====================================================================
__global__ void dft_y_kernel(const unsigned short* __restrict__ Utr,
                             const unsigned short* __restrict__ Uti,
                             const unsigned short* __restrict__ tw,
                             unsigned short* __restrict__ Are,
                             unsigned short* __restrict__ Aim) {
  const int lane = threadIdx.x & 31;
  const int plane = (int)((blockIdx.x * 256 + threadIdx.x) >> 5);  // < 16384
  const int half = lane >> 4;
  const int kx  = lane & 15;

  v16bf Ac  = load_frag(tw, 2, lane);
  v16bf As  = load_frag(tw, 3, lane);
  v16bf Asn = load_frag(tw, 4, lane);

  const unsigned short* br = Utr + (size_t)plane * 512 + kx * 32 + half * 16;
  const unsigned short* bi = Uti + (size_t)plane * 512 + kx * 32 + half * 16;
  v16bf BUr = mkbf(*(const us8*)br, *(const us8*)(br + 8));
  v16bf BUi = mkbf(*(const us8*)bi, *(const us8*)(bi + 8));

  v8f zero = {};
  // (c - i s)(ur + i ui): re = c*ur + s*ui ; im = c*ui - s*ur
  v8f xr = __builtin_amdgcn_wmma_f32_16x16x32_bf16(false, Ac,  false, BUr, (short)0, zero, false, false);
  xr     = __builtin_amdgcn_wmma_f32_16x16x32_bf16(false, As,  false, BUi, (short)0, xr,   false, false);
  v8f xi = __builtin_amdgcn_wmma_f32_16x16x32_bf16(false, Ac,  false, BUi, (short)0, zero, false, false);
  xi     = __builtin_amdgcn_wmma_f32_16x16x32_bf16(false, Asn, false, BUr, (short)0, xi,   false, false);

  const int bs = plane >> 6, c = plane & 63;
#pragma unroll
  for (int rr = 0; rr < 8; ++rr) {
    int ky = rr + 8 * half;
    size_t off = ((size_t)(ky * 16 + kx) * BS + bs) * CC + c;
    Are[off] = f2bf(xr[rr]); Aim[off] = f2bf(xi[rr]);
  }
}

// =====================================================================
// Kernel 2: repack spectral weights -> mode-major, N-major bf16 [256][nout][Cin]
// =====================================================================
__global__ void repack_w_kernel(const float* __restrict__ wr, const float* __restrict__ wi,
                                unsigned short* __restrict__ Bre,
                                unsigned short* __restrict__ Bim, int nout) {
  size_t id = (size_t)blockIdx.x * 256 + threadIdx.x;
  size_t per_mode = (size_t)CC * nout;
  int mode = (int)(id / per_mode);
  size_t rem = id - (size_t)mode * per_mode;
  int o = (int)(rem / CC), i = (int)(rem % CC);
  size_t s = ((size_t)i * nout + o) * NMODE + mode;
  Bre[id] = f2bf(wr[s]);
  Bim[id] = f2bf(wi[s]);
}

// =====================================================================
// Kernel 3: per-mode complex GEMM (WMMA) -> transposed bf16 modes
//   Xt[plane = bs*N + o][kx*16 + ky]
// =====================================================================
__global__ void mode_mix_kernel(const unsigned short* __restrict__ Are,
                                const unsigned short* __restrict__ Aim,
                                const unsigned short* __restrict__ Bre,
                                const unsigned short* __restrict__ Bim,
                                unsigned short* __restrict__ Xre,
                                unsigned short* __restrict__ Xim, int N) {
  const int lane = threadIdx.x & 31;
  const int wid  = (int)((blockIdx.x * blockDim.x + threadIdx.x) >> 5);
  const int ntiles = N >> 4;
  const int tpm = ntiles << 4;
  const int mode = wid / tpm;
  const int r = wid - mode * tpm;
  const int mt = r / ntiles, nt = r - mt * ntiles;
  const int half = lane >> 4;
  const int mrow = mt * 16 + (lane & 15);
  const int ncol = nt * 16 + (lane & 15);

  const unsigned short* Arp = Are + ((size_t)mode * BS + mrow) * CC;
  const unsigned short* Aip = Aim + ((size_t)mode * BS + mrow) * CC;
  const unsigned short* Brp = Bre + ((size_t)mode * N + ncol) * CC;
  const unsigned short* Bip = Bim + ((size_t)mode * N + ncol) * CC;

  const us8 sgn = {0x8000, 0x8000, 0x8000, 0x8000, 0x8000, 0x8000, 0x8000, 0x8000};

  v8f cre = {}; v8f cim = {};
#pragma unroll
  for (int kk = 0; kk < CC; kk += 32) {
    us8 a0 = *(const us8*)(Arp + kk + half * 16);
    us8 a1 = *(const us8*)(Arp + kk + half * 16 + 8);
    us8 i0 = *(const us8*)(Aip + kk + half * 16);
    us8 i1 = *(const us8*)(Aip + kk + half * 16 + 8);
    us8 b0 = *(const us8*)(Brp + kk + half * 16);
    us8 b1 = *(const us8*)(Brp + kk + half * 16 + 8);
    us8 c0 = *(const us8*)(Bip + kk + half * 16);
    us8 c1 = *(const us8*)(Bip + kk + half * 16 + 8);
    v16bf ar  = mkbf(a0, a1);
    v16bf ai  = mkbf(i0, i1);
    v16bf ain = mkbf(i0 ^ sgn, i1 ^ sgn);
    v16bf br  = mkbf(b0, b1);
    v16bf bi  = mkbf(c0, c1);
    cre = __builtin_amdgcn_wmma_f32_16x16x32_bf16(false, ar,  false, br, (short)0, cre, false, false);
    cre = __builtin_amdgcn_wmma_f32_16x16x32_bf16(false, ain, false, bi, (short)0, cre, false, false);
    cim = __builtin_amdgcn_wmma_f32_16x16x32_bf16(false, ar,  false, bi, (short)0, cim, false, false);
    cim = __builtin_amdgcn_wmma_f32_16x16x32_bf16(false, ai,  false, br, (short)0, cim, false, false);
  }
  const int kxm = mode & 15, kym = mode >> 4;
#pragma unroll
  for (int rr = 0; rr < 8; ++rr) {
    int m = mt * 16 + rr + half * 8;
    size_t plane = (size_t)m * N + ncol;
    size_t off = plane * NMODE + kxm * 16 + kym;
    Xre[off] = f2bf(cre[rr]); Xim[off] = f2bf(cim[rr]);
  }
}

// =====================================================================
// Kernel 4: inverse DFT via WMMA. 2 waves per plane, 4 planes/block.
// =====================================================================
__global__ void idft_wmma_kernel(const unsigned short* __restrict__ Xre,
                                 const unsigned short* __restrict__ Xim,
                                 const unsigned short* __restrict__ tw,
                                 const float* __restrict__ resid,
                                 float* __restrict__ out_f32,
                                 unsigned short* __restrict__ out_bf, int N) {
  __shared__ unsigned short ldsTr[8][256];
  __shared__ unsigned short ldsTi[8][256];
  const int tid = threadIdx.x;
  const int lane = tid & 31;
  const int w = tid >> 5;
  const int half = lane >> 4;
  const int p = blockIdx.x * 4 + (w >> 1);
  const int mh = w & 1;                    // y-half
  const int y0 = mh << 4;
  const int bs = p / N, o = p % N;

  // ---- stage I1: T = E * X  (K = ky padded 16->32)
  v16bf Ec  = load_frag(tw, 5 + mh * 3, lane);
  v16bf Es  = load_frag(tw, 6 + mh * 3, lane);
  v16bf Esn = load_frag(tw, 7 + mh * 3, lane);

  const int kx = lane & 15;
  us8 z8 = {0, 0, 0, 0, 0, 0, 0, 0};
  us8 xr0 = z8, xr1 = z8, xi0 = z8, xi1 = z8;
  if (half == 0) {
    const unsigned short* bp = Xre + (size_t)p * NMODE + kx * 16;
    const unsigned short* bq = Xim + (size_t)p * NMODE + kx * 16;
    xr0 = *(const us8*)bp; xr1 = *(const us8*)(bp + 8);
    xi0 = *(const us8*)bq; xi1 = *(const us8*)(bq + 8);
  }
  v16bf BXr = mkbf(xr0, xr1), BXi = mkbf(xi0, xi1);
  v8f zero = {};
  // (c + i s)(Xr + i Xi): Tr = c*Xr - s*Xi ; Ti = c*Xi + s*Xr
  v8f tr = __builtin_amdgcn_wmma_f32_16x16x32_bf16(false, Ec,  false, BXr, (short)0, zero, false, false);
  tr     = __builtin_amdgcn_wmma_f32_16x16x32_bf16(false, Esn, false, BXi, (short)0, tr,   false, false);
  v8f ti = __builtin_amdgcn_wmma_f32_16x16x32_bf16(false, Ec,  false, BXi, (short)0, zero, false, false);
  ti     = __builtin_amdgcn_wmma_f32_16x16x32_bf16(false, Es,  false, BXr, (short)0, ti,   false, false);
#pragma unroll
  for (int rr = 0; rr < 8; ++rr) {
    int yl = rr + 8 * half;
    ldsTr[w][yl * 16 + kx] = f2bf(tr[rr]);
    ldsTi[w][yl * 16 + kx] = f2bf(ti[rr]);
  }
  // same-wave LDS RAW: DS ops are in-order per wave; compiler inserts waits.

  // ---- stage I2: out = Tr*Cc + Ti*Csn  (K = kx padded)
  const int yl2 = lane & 15;
  v16bf ATr, ATi;
#pragma unroll
  for (int j = 0; j < 8; ++j) {
    int kp = a_kpos(j, half);
#pragma unroll
    for (int e = 0; e < 2; ++e) {
      int K = kp + e;
      unsigned short vr = 0, vi = 0;
      if (K < MD) { vr = ldsTr[w][yl2 * 16 + K]; vi = ldsTi[w][yl2 * 16 + K]; }
      ATr[2 * j + e] = bfbits(vr);
      ATi[2 * j + e] = bfbits(vi);
    }
  }
#pragma unroll
  for (int nt = 0; nt < 2; ++nt) {
    int x = nt * 16 + (lane & 15);
    v16bf Bc  = load_frag(tw, 11 + nt * 2, lane);
    v16bf Bsn = load_frag(tw, 12 + nt * 2, lane);
    v8f acc = __builtin_amdgcn_wmma_f32_16x16x32_bf16(false, ATr, false, Bc,  (short)0, zero, false, false);
    acc     = __builtin_amdgcn_wmma_f32_16x16x32_bf16(false, ATi, false, Bsn, (short)0, acc,  false, false);
#pragma unroll
    for (int rr = 0; rr < 8; ++rr) {
      int y = y0 + rr + 8 * half;
      int pix = y * 32 + x;
      float v = acc[rr];
      if (resid) v += resid[((size_t)bs * CC + (o & 63)) * PIX + pix];
      if (out_bf) out_bf[(size_t)p * PIX + pix] = f2bf(v);
      else        out_f32[(size_t)p * PIX + pix] = v;
    }
  }
}

// =====================================================================
// Kernel 5: Swin-v2 log-CPB bias MLP -> bias[head][si][sj]
// =====================================================================
__global__ void cpb_kernel(const float* __restrict__ w1, const float* __restrict__ b1,
                           const float* __restrict__ w2, float* __restrict__ bias) {
  int id = blockIdx.x * 256 + threadIdx.x;       // < 4096
  int si = id >> 6, sj = id & 63;
  float dy = (float)(si >> 3) - (float)(sj >> 3);
  float dx = (float)(si & 7)  - (float)(sj & 7);
  float sy = (dy > 0.f) ? 1.f : ((dy < 0.f) ? -1.f : 0.f);
  float sx = (dx > 0.f) ? 1.f : ((dx < 0.f) ? -1.f : 0.f);
  dy = sy * log1pf(fabsf(dy));
  dx = sx * log1pf(fabsf(dx));
  float acc[NH];
#pragma unroll
  for (int hh = 0; hh < NH; ++hh) acc[hh] = 0.f;
  for (int c = 0; c < CC; ++c) {
    float hv = dy * w1[c] + dx * w1[CC + c] + b1[c];
    hv = fmaxf(hv, 0.f);
#pragma unroll
    for (int hh = 0; hh < NH; ++hh) acc[hh] += hv * w2[c * NH + hh];
  }
#pragma unroll
  for (int hh = 0; hh < NH; ++hh) bias[((size_t)hh * 64 + si) * 64 + sj] = acc[hh];
}

// =====================================================================
// Kernel 6: attention scores via WMMA: per (b,h) GEMM 64x64, K=8192
// =====================================================================
__global__ void scores_kernel(const unsigned short* __restrict__ qkv,
                              float* __restrict__ scores) {
  const int lane = threadIdx.x & 31;
  const int wid  = (int)((blockIdx.x * 256 + threadIdx.x) >> 5);
  const int bh = wid >> 4;
  const int r = wid & 15;
  const int mt = r >> 2, nt = r & 3;
  const int b = bh >> 3, h = bh & 7;
  const int half = lane >> 4;
  const int si = mt * 16 + (lane & 15);
  const int sj = nt * 16 + (lane & 15);

  const unsigned short* qbase = qkv + ((size_t)(b * SS + si) * NQKV + h * HD) * PIX;
  const unsigned short* kbase = qkv + ((size_t)(b * SS + sj) * NQKV + CC + h * HD) * PIX;

  v8f acc = {};
  for (int kk = 0; kk < KATT; kk += 32) {
    us8 a0 = *(const us8*)(qbase + kk + half * 16);
    us8 a1 = *(const us8*)(qbase + kk + half * 16 + 8);
    us8 b0 = *(const us8*)(kbase + kk + half * 16);
    us8 b1 = *(const us8*)(kbase + kk + half * 16 + 8);
    acc = __builtin_amdgcn_wmma_f32_16x16x32_bf16(false, mkbf(a0, a1), false, mkbf(b0, b1),
                                                  (short)0, acc, false, false);
  }
  const float scale = 1.0f / (2.82842712474619f * 1024.0f);  // 1/(sqrt(8)*H*W)
#pragma unroll
  for (int rr = 0; rr < 8; ++rr) {
    int m = mt * 16 + rr + half * 8;
    scores[((size_t)(b * NH + h) * 64 + m) * 64 + nt * 16 + (lane & 15)] = acc[rr] * scale;
  }
}

// =====================================================================
// Kernel 7: softmax over sj with bias -> bf16 attention weights
// =====================================================================
__global__ void softmax_kernel(const float* __restrict__ scores, const float* __restrict__ bias,
                               unsigned short* __restrict__ attn_bf) {
  int row = blockIdx.x * 256 + threadIdx.x;   // < 2048
  int b = row >> 9, rem = row & 511, h = rem >> 6, si = rem & 63;
  const float* sp = scores + ((size_t)(b * NH + h) * 64 + si) * 64;
  const float* bp = bias + ((size_t)h * 64 + si) * 64;
  unsigned short* op = attn_bf + ((size_t)(b * NH + h) * 64 + si) * 64;
  float m = -3.0e38f;
  for (int j = 0; j < 64; ++j) m = fmaxf(m, sp[j] + bp[j]);
  float sum = 0.f;
  for (int j = 0; j < 64; ++j) sum += __expf(sp[j] + bp[j] - m);
  float rs = 1.0f / sum;
  for (int j = 0; j < 64; ++j) op[j] = f2bf(__expf(sp[j] + bp[j] - m) * rs);
}

// =====================================================================
// Kernel 8: sa = attn @ v via WMMA; attn bf16, sa stored bf16
// =====================================================================
__global__ void attnv_kernel(const unsigned short* __restrict__ attn_bf,
                             const unsigned short* __restrict__ qkv,
                             unsigned short* __restrict__ sa_bf) {
  const int lane = threadIdx.x & 31;
  const int wid  = (int)((blockIdx.x * 256 + threadIdx.x) >> 5);
  const int bh = wid >> 11;
  const int r = wid & 2047;
  const int mt = r >> 9, nt = r & 511;
  const int b = bh >> 3, h = bh & 7;
  const int half = lane >> 4;
  const int si = mt * 16 + (lane & 15);
  const int n = nt * 16 + (lane & 15);      // flat (d,pix)

  const unsigned short* ap = attn_bf + ((size_t)(b * NH + h) * 64 + si) * 64;
  const unsigned short* vbase =
      qkv + ((size_t)(b * SS) * NQKV + 2 * CC + h * HD) * PIX + n;

  v8f acc = {};
#pragma unroll
  for (int kk = 0; kk < 64; kk += 32) {
    v16bf a = mkbf(*(const us8*)(ap + kk + half * 16),
                   *(const us8*)(ap + kk + half * 16 + 8));
    v16bf bm;
#pragma unroll
    for (int j = 0; j < 8; ++j) {
      int kb = half * 16 + 2 * j;
#pragma unroll
      for (int e = 0; e < 2; ++e) {
        int sj = kk + kb + e;
        bm[2 * j + e] = bfbits(vbase[(size_t)sj * NQKV * PIX]);
      }
    }
    acc = __builtin_amdgcn_wmma_f32_16x16x32_bf16(false, a, false, bm, (short)0, acc, false, false);
  }
#pragma unroll
  for (int rr = 0; rr < 8; ++rr) {
    int m = mt * 16 + rr + half * 8;
    sa_bf[((size_t)(b * SS + m) * CC + h * HD) * PIX + n] = f2bf(acc[rr]);
  }
}

// =====================================================================
// host-side orchestration
// =====================================================================
extern "C" void kernel_launch(void* const* d_in, const int* in_sizes, int n_in,
                              void* d_out, int out_size, void* d_ws, size_t ws_size,
                              hipStream_t stream) {
  (void)in_sizes; (void)n_in; (void)out_size; (void)ws_size;
  const float* seq    = (const float*)d_in[0];
  const float* qkv_wr = (const float*)d_in[1];
  const float* qkv_wi = (const float*)d_in[2];
  const float* out_wr = (const float*)d_in[3];
  const float* out_wi = (const float*)d_in[4];
  const float* cpb_w1 = (const float*)d_in[5];
  const float* cpb_b1 = (const float*)d_in[6];
  const float* cpb_w2 = (const float*)d_in[7];

  char* ws = (char*)d_ws;
  size_t off = 0;
  auto alloc = [&](size_t bytes) -> void* {
    void* p = ws + off;
    off += (bytes + 255) & ~(size_t)255;
    return p;
  };
  unsigned short* tw     = (unsigned short*)alloc((size_t)NFRAG * 512 * 2);
  unsigned short* seq_bf = (unsigned short*)alloc((size_t)BS * CC * PIX * 2);   // reused for sa
  unsigned short* Utr    = (unsigned short*)alloc((size_t)BS * CC * 512 * 2);
  unsigned short* Uti    = (unsigned short*)alloc((size_t)BS * CC * 512 * 2);
  unsigned short* XfAre  = (unsigned short*)alloc((size_t)NMODE * BS * CC * 2);
  unsigned short* XfAim  = (unsigned short*)alloc((size_t)NMODE * BS * CC * 2);
  unsigned short* WBre   = (unsigned short*)alloc((size_t)NMODE * CC * NQKV * 2);
  unsigned short* WBim   = (unsigned short*)alloc((size_t)NMODE * CC * NQKV * 2);
  unsigned short* Xtre   = (unsigned short*)alloc((size_t)BS * NQKV * NMODE * 2);
  unsigned short* Xtim   = (unsigned short*)alloc((size_t)BS * NQKV * NMODE * 2);
  unsigned short* qkvs   = (unsigned short*)alloc((size_t)BS * NQKV * PIX * 2);
  float* bias = (float*)alloc((size_t)NH * 64 * 64 * 4);
  float* scr  = (float*)alloc((size_t)BB * NH * 64 * 64 * 4);
  unsigned short* attn_bf = (unsigned short*)alloc((size_t)BB * NH * 64 * 64 * 2);
  unsigned short* sa_bf = seq_bf;   // reuse (seq_bf dead after qkv is built)

  // 0) twiddle fragment table
  twiddle_init_kernel<<<(NFRAG * 512 + 255) / 256, 256, 0, stream>>>(tw);
  // 1) seq -> bf16
  cvt_bf16_kernel<<<(BS * CC * PIX) / 256, 256, 0, stream>>>(seq, seq_bf);
  // 2) forward transforms (WMMA)
  dft_x_kernel<<<(BS * CC * 2) / 8, 256, 0, stream>>>(seq_bf, tw, Utr, Uti);
  dft_y_kernel<<<(BS * CC) / 8, 256, 0, stream>>>(Utr, Uti, tw, XfAre, XfAim);
  // 3) repack qkv weights
  repack_w_kernel<<<(NMODE * CC * NQKV) / 256, 256, 0, stream>>>(qkv_wr, qkv_wi, WBre, WBim, NQKV);
  // 4) per-mode complex GEMM (WMMA)
  mode_mix_kernel<<<(NMODE * 16 * (NQKV / 16)) / 8, 256, 0, stream>>>(
      XfAre, XfAim, WBre, WBim, Xtre, Xtim, NQKV);
  // 5) inverse transform (WMMA) + tiled residual -> qkv bf16
  idft_wmma_kernel<<<(BS * NQKV) / 4, 256, 0, stream>>>(Xtre, Xtim, tw, seq,
                                                        (float*)nullptr, qkvs, NQKV);
  // 6) CPB bias
  cpb_kernel<<<16, 256, 0, stream>>>(cpb_w1, cpb_b1, cpb_w2, bias);
  // 7) attention scores (WMMA)
  scores_kernel<<<(BB * NH * 16) / 8, 256, 0, stream>>>(qkvs, scr);
  // 8) softmax + bias -> bf16 attn
  softmax_kernel<<<8, 256, 0, stream>>>(scr, bias, attn_bf);
  // 9) attn @ v (WMMA) -> sa bf16
  attnv_kernel<<<(BB * NH * 2048) / 8, 256, 0, stream>>>(attn_bf, qkvs, sa_bf);
  // 10) forward transforms of sa (WMMA)
  dft_x_kernel<<<(BS * CC * 2) / 8, 256, 0, stream>>>(sa_bf, tw, Utr, Uti);
  dft_y_kernel<<<(BS * CC) / 8, 256, 0, stream>>>(Utr, Uti, tw, XfAre, XfAim);
  // 11) repack out weights
  repack_w_kernel<<<(NMODE * CC * CC) / 256, 256, 0, stream>>>(out_wr, out_wi, WBre, WBim, CC);
  // 12) per-mode complex GEMM (WMMA)
  mode_mix_kernel<<<(NMODE * 16 * (CC / 16)) / 8, 256, 0, stream>>>(
      XfAre, XfAim, WBre, WBim, Xtre, Xtim, CC);
  // 13) inverse transform (WMMA) -> d_out f32
  idft_wmma_kernel<<<(BS * CC) / 4, 256, 0, stream>>>(Xtre, Xtim, tw, (const float*)nullptr,
                                                      (float*)d_out, (unsigned short*)nullptr, CC);
}
